// GAT_31121333027360
// MI455X (gfx1250) — matrix operand order repaired and checked by reading, not compiled, not run on hardware
//
#include <hip/hip_runtime.h>
#include <hip/hip_bf16.h>

#define NEG_SLOPE 0.2f

typedef __attribute__((ext_vector_type(16))) __bf16 v16bf;
typedef __attribute__((ext_vector_type(8)))  __bf16 v8bf;
typedef __attribute__((ext_vector_type(8)))  float  v8f;
typedef int v4i_ __attribute__((vector_size(16)));   // pointee type expected by async-LDS builtin

union ABFrag { v16bf v; v8bf h[2]; };

__device__ __forceinline__ void atomicMaxF(float* addr, float val) {
  if (val >= 0.0f) atomicMax((int*)addr, __float_as_int(val));
  else             atomicMin((unsigned int*)addr, (unsigned int)__float_as_int(val));
}

// 16B global -> LDS copy; CDNA5 async path (ASYNCcnt) when the builtin exists.
__device__ __forceinline__ void copy_b128_to_lds(void* lds, const void* g) {
#if __has_builtin(__builtin_amdgcn_global_load_async_to_lds_b128)
  __builtin_amdgcn_global_load_async_to_lds_b128(
      (v4i_*)const_cast<void*>(g), (v4i_*)lds, 0, 0);
#else
  *(uint4*)lds = *(const uint4*)g;
#endif
}

__device__ __forceinline__ void wait_lds_copies() {
#if __has_builtin(__builtin_amdgcn_s_wait_asynccnt)
  __builtin_amdgcn_s_wait_asynccnt(0);
#endif
}

// ---------------- elementwise ----------------
__global__ void k_f32_to_bf16(const float* __restrict__ in, __bf16* __restrict__ out, long long n) {
  long long i = (long long)blockIdx.x * blockDim.x + threadIdx.x;
  if (i < n) out[i] = (__bf16)in[i];
}

// convert + transpose: in is [K][Nn] row-major, out is [Nn][K] row-major
__global__ void k_f32_to_bf16_T(const float* __restrict__ in, __bf16* __restrict__ out,
                                int K, int Nn) {
  long long i = (long long)blockIdx.x * blockDim.x + threadIdx.x;
  if (i < (long long)K * Nn) {
    int k = (int)(i / Nn), n = (int)(i % Nn);
    out[(size_t)n * K + k] = (__bf16)in[i];
  }
}

__global__ void k_elu_to_bf16(const float* __restrict__ in, __bf16* __restrict__ out, long long n) {
  long long i = (long long)blockIdx.x * blockDim.x + threadIdx.x;
  if (i < n) {
    float x = in[i];
    out[i] = (__bf16)(x > 0.0f ? x : (__expf(x) - 1.0f));
  }
}

__global__ void k_fill_f32(float* __restrict__ p, float v, long long n) {
  long long i = (long long)blockIdx.x * blockDim.x + threadIdx.x;
  if (i < n) p[i] = v;
}

__global__ void k_fill_bias(float* __restrict__ p, const float* __restrict__ b, int mask, long long n) {
  long long i = (long long)blockIdx.x * blockDim.x + threadIdx.x;
  if (i < n) p[i] = b[(int)(i & (long long)mask)];
}

// ---------------- WMMA GEMM: C[M,N] = A[M,K](bf16, row-major) * Bt[N,K](bf16, row-major) ----------------
template<int BM, int BN, int WR, int WC>
__global__ __launch_bounds__(256) void k_gemm_bf16(
    const __bf16* __restrict__ A, const __bf16* __restrict__ Bt,
    float* __restrict__ C, int M, int Nn, int K) {
  constexpr int MT = BM / (16 * WR);   // WMMA tiles per wave (M dir)
  constexpr int NT = BN / (16 * WC);   // WMMA tiles per wave (N dir)
  __shared__ alignas(16) __bf16 lA[BM][40];   // [m][k] tile (pad keeps 16B row align)
  __shared__ alignas(16) __bf16 lB[BN][40];   // [n][k] tile

  const int tid   = threadIdx.x;
  const int lane  = tid & 31;
  const int wave  = tid >> 5;
  const int wr    = wave % WR;
  const int wc    = wave / WR;
  const int l16   = lane & 15;
  const int lhalf = lane >> 4;

  const int mBlock = blockIdx.y * BM;
  const int nBlock = blockIdx.x * BN;
  const int mWave  = wr * MT * 16;
  const int nWave  = wc * NT * 16;

  v8f acc[MT][NT] = {};

  for (int kt = 0; kt < K; kt += 32) {
    // --- stage A tile: BM x 32 as 16B chunks (async to LDS on CDNA5) ---
    constexpr int AV = (BM * 4) / 256;     // 4 chunks per 32-wide row
    #pragma unroll
    for (int j = 0; j < AV; ++j) {
      int v  = tid + j * 256;
      int m  = v >> 2;
      int kc = (v & 3) * 8;
      int row = mBlock + m;
      if (row < M) {
        copy_b128_to_lds(&lA[m][kc], A + (size_t)row * K + kt + kc);
      } else {
        uint4 z = {0u, 0u, 0u, 0u};
        *(uint4*)(&lA[m][kc]) = z;
      }
    }
    // --- stage B tile: BN x 32 as 16B chunks from pre-transposed Bt ---
    constexpr int BV = (BN * 4) / 256;
    #pragma unroll
    for (int j = 0; j < BV; ++j) {
      int v  = tid + j * 256;
      int n  = v >> 2;
      int kc = (v & 3) * 8;
      copy_b128_to_lds(&lB[n][kc], Bt + (size_t)(nBlock + n) * K + kt + kc);
    }
    wait_lds_copies();
    __syncthreads();

    // --- fragments per documented CDNA5 16-bit layouts ---
    ABFrag af[MT], bfr[NT];
    #pragma unroll
    for (int mi = 0; mi < MT; ++mi) {
      // A 16x32: low lanes K 0-7 & 16-23, high lanes K 8-15 & 24-31; row M = lane&15
      const __bf16* ap = &lA[mWave + mi * 16 + l16][lhalf * 8];
      af[mi].h[0] = *(const v8bf*)ap;
      af[mi].h[1] = *(const v8bf*)(ap + 16);
    }
    #pragma unroll
    for (int ni = 0; ni < NT; ++ni) {
      // B 32x16: col N = lane&15; low lanes K 0-7 & 16-23, high lanes K 8-15 & 24-31
      const __bf16* bp = &lB[nWave + ni * 16 + l16][lhalf * 8];
      bfr[ni].h[0] = *(const v8bf*)bp;
      bfr[ni].h[1] = *(const v8bf*)(bp + 16);
    }
    #pragma unroll
    for (int mi = 0; mi < MT; ++mi)
      #pragma unroll
      for (int ni = 0; ni < NT; ++ni)
        acc[mi][ni] = __builtin_amdgcn_wmma_f32_16x16x32_bf16(
            false, af[mi].v, false, bfr[ni].v, (short)0, acc[mi][ni], false, false);
    __syncthreads();
  }

  // --- epilogue: VGPR i -> M=i (low lanes) / M=8+i (high lanes), N=lane&15 ---
  #pragma unroll
  for (int mi = 0; mi < MT; ++mi)
    #pragma unroll
    for (int ni = 0; ni < NT; ++ni)
      #pragma unroll
      for (int i = 0; i < 8; ++i) {
        int row = mBlock + mWave + mi * 16 + (lhalf ? i + 8 : i);
        int col = nBlock + nWave + ni * 16 + l16;
        if (row < M) C[(size_t)row * Nn + col] = acc[mi][ni][i];
      }
}

// ---------------- attention scores: one wave per (node, head) ----------------
template<int H, int C>
__global__ __launch_bounds__(256) void k_att_scores(
    const float* __restrict__ h, const float* __restrict__ att_s,
    const float* __restrict__ att_d, float* __restrict__ as,
    float* __restrict__ ad, int N) {
  int pair = blockIdx.x * (blockDim.x >> 5) + (threadIdx.x >> 5);
  int lane = threadIdx.x & 31;
  if (pair >= N * H) return;
  int n = pair / H, hh = pair % H;
  const float* hp = h + (size_t)n * (H * C) + hh * C;
  float ss = 0.0f, sd = 0.0f;
  for (int c = lane; c < C; c += 32) {
    float v = hp[c];
    ss += v * att_s[hh * C + c];
    sd += v * att_d[hh * C + c];
  }
  #pragma unroll
  for (int off = 16; off > 0; off >>= 1) {
    ss += __shfl_xor(ss, off);
    sd += __shfl_xor(sd, off);
  }
  if (lane == 0) { as[pair] = ss; ad[pair] = sd; }
}

// ---------------- edge passes (edge id >= E -> self loop) ----------------
template<int H>
__global__ __launch_bounds__(256) void k_edge_max(
    const int* __restrict__ ei, const float* __restrict__ as,
    const float* __restrict__ ad, float* __restrict__ emax, int E, int N) {
  long long gid = (long long)blockIdx.x * blockDim.x + threadIdx.x;
  if (gid >= (long long)(E + N) * H) return;
  int e = (int)(gid / H), hh = (int)(gid % H);
  int s, d;
  if (e < E) { s = ei[e]; d = ei[E + e]; } else { s = d = e - E; }
  float v = as[s * H + hh] + ad[d * H + hh];
  v = v > 0.0f ? v : NEG_SLOPE * v;
  atomicMaxF(&emax[d * H + hh], v);
}

template<int H>
__global__ __launch_bounds__(256) void k_edge_sum(
    const int* __restrict__ ei, const float* __restrict__ as,
    const float* __restrict__ ad, const float* __restrict__ emax,
    float* __restrict__ den, int E, int N) {
  long long gid = (long long)blockIdx.x * blockDim.x + threadIdx.x;
  if (gid >= (long long)(E + N) * H) return;
  int e = (int)(gid / H), hh = (int)(gid % H);
  int s, d;
  if (e < E) { s = ei[e]; d = ei[E + e]; } else { s = d = e - E; }
  float v = as[s * H + hh] + ad[d * H + hh];
  v = v > 0.0f ? v : NEG_SLOPE * v;
  atomicAdd(&den[d * H + hh], __expf(v - emax[d * H + hh]));
}

template<int H, int C>
__global__ __launch_bounds__(256) void k_edge_aggr(
    const int* __restrict__ ei, const float* __restrict__ as,
    const float* __restrict__ ad, const float* __restrict__ emax,
    const float* __restrict__ den, const float* __restrict__ h,
    float* __restrict__ out, int E, int N) {
  constexpr int CH = (H * C) / 4;     // float4 chunks per edge
  long long gid = (long long)blockIdx.x * blockDim.x + threadIdx.x;
  if (gid >= (long long)(E + N) * CH) return;
  int e  = (int)(gid / CH);
  int ch = (int)(gid % CH);
  int c4 = ch * 4;
  int hh = c4 / C;
  int s, d;
  if (e < E) { s = ei[e]; d = ei[E + e]; } else { s = d = e - E; }
  float v = as[s * H + hh] + ad[d * H + hh];
  v = v > 0.0f ? v : NEG_SLOPE * v;
  float alpha = __expf(v - emax[d * H + hh]) / (den[d * H + hh] + 1e-16f);
  const float4 hv = *(const float4*)(h + (size_t)s * (H * C) + c4);
  float* op = out + (size_t)d * (H * C) + c4;
  atomicAdd(op + 0, alpha * hv.x);
  atomicAdd(op + 1, alpha * hv.y);
  atomicAdd(op + 2, alpha * hv.z);
  atomicAdd(op + 3, alpha * hv.w);
}

// ---------------- host ----------------
extern "C" void kernel_launch(void* const* d_in, const int* in_sizes, int n_in,
                              void* d_out, int out_size, void* d_ws, size_t ws_size,
                              hipStream_t stream) {
  constexpr int N = 10000, E = 160000, F = 256;
  constexpr int H1 = 8, C1 = 128, HC1 = H1 * C1;   // 1024
  constexpr int H2 = 1, C2 = 64,  HC2 = H2 * C2;   // 64

  const float* x      = (const float*)d_in[0];
  const int*   ei     = (const int*)d_in[1];       // [2, E] int32
  const float* W1     = (const float*)d_in[2];
  const float* att_s1 = (const float*)d_in[3];
  const float* att_d1 = (const float*)d_in[4];
  const float* b1     = (const float*)d_in[5];
  const float* W2     = (const float*)d_in[6];
  const float* att_s2 = (const float*)d_in[7];
  const float* att_d2 = (const float*)d_in[8];
  const float* b2     = (const float*)d_in[9];
  float* out = (float*)d_out;

  char* base = (char*)d_ws;
  size_t off = 0;
  auto carve = [&](size_t bytes) -> void* {
    void* p = base + off;
    off += (bytes + 255) & ~(size_t)255;
    return p;
  };
  __bf16* xb    = (__bf16*)carve((size_t)N * F * 2);
  __bf16* w1t   = (__bf16*)carve((size_t)F * HC1 * 2);   // [HC1][F]
  __bf16* w2t   = (__bf16*)carve((size_t)HC1 * HC2 * 2); // [HC2][HC1]
  float*  h1    = (float*)carve((size_t)N * HC1 * 4);
  float*  out1  = (float*)carve((size_t)N * HC1 * 4);
  __bf16* heb   = (__bf16*)carve((size_t)N * HC1 * 2);
  float*  h2    = (float*)carve((size_t)N * HC2 * 4);
  float*  as1   = (float*)carve((size_t)N * H1 * 4);
  float*  ad1   = (float*)carve((size_t)N * H1 * 4);
  float*  emax1 = (float*)carve((size_t)N * H1 * 4);
  float*  den1  = (float*)carve((size_t)N * H1 * 4);
  float*  as2   = (float*)carve((size_t)N * H2 * 4);
  float*  ad2   = (float*)carve((size_t)N * H2 * 4);
  float*  emax2 = (float*)carve((size_t)N * H2 * 4);
  float*  den2  = (float*)carve((size_t)N * H2 * 4);

  auto cdiv = [](long long a, long long b) -> unsigned { return (unsigned)((a + b - 1) / b); };
  const float NEG_INF = -__builtin_huge_valf();

  // ---- conversions (weights transposed to [N][K] for contiguous LDS staging) ----
  k_f32_to_bf16<<<cdiv((long long)N * F, 256), 256, 0, stream>>>(x, xb, (long long)N * F);
  k_f32_to_bf16_T<<<cdiv((long long)F * HC1, 256), 256, 0, stream>>>(W1, w1t, F, HC1);
  k_f32_to_bf16_T<<<cdiv((long long)HC1 * HC2, 256), 256, 0, stream>>>(W2, w2t, HC1, HC2);

  // ---- layer 1: h1 = x @ W1 (WMMA bf16) ----
  k_gemm_bf16<128, 128, 4, 2><<<dim3(HC1 / 128, cdiv(N, 128)), 256, 0, stream>>>(
      xb, w1t, h1, N, HC1, F);

  k_att_scores<H1, C1><<<cdiv((long long)N * H1, 8), 256, 0, stream>>>(
      h1, att_s1, att_d1, as1, ad1, N);

  k_fill_f32<<<cdiv((long long)N * H1, 256), 256, 0, stream>>>(emax1, NEG_INF, (long long)N * H1);
  k_fill_f32<<<cdiv((long long)N * H1, 256), 256, 0, stream>>>(den1, 0.0f, (long long)N * H1);
  k_fill_bias<<<cdiv((long long)N * HC1, 256), 256, 0, stream>>>(out1, b1, HC1 - 1, (long long)N * HC1);

  k_edge_max<H1><<<cdiv((long long)(E + N) * H1, 256), 256, 0, stream>>>(ei, as1, ad1, emax1, E, N);
  k_edge_sum<H1><<<cdiv((long long)(E + N) * H1, 256), 256, 0, stream>>>(ei, as1, ad1, emax1, den1, E, N);
  k_edge_aggr<H1, C1><<<cdiv((long long)(E + N) * (HC1 / 4), 256), 256, 0, stream>>>(
      ei, as1, ad1, emax1, den1, h1, out1, E, N);

  // ---- ELU + bf16 for layer-2 GEMM ----
  k_elu_to_bf16<<<cdiv((long long)N * HC1, 256), 256, 0, stream>>>(out1, heb, (long long)N * HC1);

  // ---- layer 2: h2 = elu(out1) @ W2 (WMMA bf16) ----
  k_gemm_bf16<128, 64, 8, 1><<<dim3(HC2 / 64, cdiv(N, 128)), 256, 0, stream>>>(
      heb, w2t, h2, N, HC2, HC1);

  k_att_scores<H2, C2><<<cdiv((long long)N * H2, 8), 256, 0, stream>>>(
      h2, att_s2, att_d2, as2, ad2, N);

  k_fill_f32<<<cdiv((long long)N * H2, 256), 256, 0, stream>>>(emax2, NEG_INF, (long long)N * H2);
  k_fill_f32<<<cdiv((long long)N * H2, 256), 256, 0, stream>>>(den2, 0.0f, (long long)N * H2);
  k_fill_bias<<<cdiv((long long)N * HC2, 256), 256, 0, stream>>>(out, b2, HC2 - 1, (long long)N * HC2);

  k_edge_max<H2><<<cdiv((long long)(E + N) * H2, 256), 256, 0, stream>>>(ei, as2, ad2, emax2, E, N);
  k_edge_sum<H2><<<cdiv((long long)(E + N) * H2, 256), 256, 0, stream>>>(ei, as2, ad2, emax2, den2, E, N);
  k_edge_aggr<H2, C2><<<cdiv((long long)(E + N) * (HC2 / 4), 256), 256, 0, stream>>>(
      ei, as2, ad2, emax2, den2, h2, out, E, N);
}
